// Attention_4_lora_24180665876844
// MI455X (gfx1250) — matrix operand order repaired and checked by reading, not compiled, not run on hardware
//
#include <hip/hip_runtime.h>
#include <hip/hip_bf16.h>
#include <math.h>

// ---------------------------------------------------------------------------
// Problem constants (from reference): B=8, T=1024, C=1024, R=64
// ---------------------------------------------------------------------------
#define Bsz   8
#define Tsz   1024
#define Csz   1024
#define Rsz   64
#define C3    (3 * Csz)
#define ROWS  (Bsz * Tsz)          // 8192
#define BN_EPS 1e-5f

// gfx1250 async global->LDS staging (ASYNCcnt-tracked). Set to 0 to fall back
// to sync load + ds_store if the assembler rejects the mnemonic.
#define USE_ASYNC_LDS 1

typedef __attribute__((ext_vector_type(16))) __bf16 v16bf;
typedef __attribute__((ext_vector_type(8)))  float  v8f;

// ---------------------------------------------------------------------------
// 16B global -> LDS copy: async (direct-to-LDS DMA path) or sync fallback.
// Generic LDS pointer low 32 bits == LDS byte address (ISA flat aperture rule).
// ---------------------------------------------------------------------------
__device__ __forceinline__ void copy16_g2l(const __bf16* g, __bf16* l) {
#if USE_ASYNC_LDS
    unsigned      lds  = (unsigned)(unsigned long long)(uintptr_t)l;
    unsigned long long ga = (unsigned long long)(uintptr_t)g;
    asm volatile("global_load_async_to_lds_b128 %0, %1, off"
                 :: "v"(lds), "v"(ga) : "memory");
#else
    *(uint4*)l = *(const uint4*)g;
#endif
}

__device__ __forceinline__ void wait_g2l() {
#if USE_ASYNC_LDS
    asm volatile("s_wait_asynccnt 0x0" ::: "memory");
#endif
}

// ---------------------------------------------------------------------------
// Utility: f32 -> bf16 convert
// ---------------------------------------------------------------------------
__global__ __launch_bounds__(256) void cvt_f32_bf16(const float* __restrict__ in,
                                                    __bf16* __restrict__ out, int n) {
    int i = blockIdx.x * 256 + threadIdx.x;
    if (i < n) out[i] = (__bf16)in[i];
}

// ---------------------------------------------------------------------------
// LoRA merge: out[i] = bf16( W[i] + sum_r LA[i/nB, r] * LB[r, i%nB] )
// ---------------------------------------------------------------------------
__global__ __launch_bounds__(256) void lora_merge(const float* __restrict__ W,
                                                  const float* __restrict__ LA,
                                                  const float* __restrict__ LB,
                                                  __bf16* __restrict__ out,
                                                  int total, int nB) {
    int i = blockIdx.x * 256 + threadIdx.x;
    if (i >= total) return;
    int r0 = i / nB;
    int c0 = i - r0 * nB;
    const float* a = LA + (size_t)r0 * Rsz;
    float acc = W[i];
#pragma unroll 8
    for (int r = 0; r < Rsz; ++r) acc += a[r] * LB[(size_t)r * nB + c0];
    out[i] = (__bf16)acc;
}

// ---------------------------------------------------------------------------
// WMMA bf16 GEMM:  C[m,n] = sum_k A[m,k] * Wt[n,k]   (both row-major, K inner)
//   128x128 C-tile per 256-thread block (8 wave32).
//   Wave w: tile rows 2*(w&3)..+1, tile cols 4*(w>>2)..+3  -> 2x4 v8f accums,
//   8 wmma per 32-wide K slab from 2 A-frags x 4 B-frags.
//   K slabs double-buffered in LDS, staged with async global->LDS (ASYNCcnt),
//   one barrier per slab. Padded LDS stride (40 bf16 = 80B) kills conflicts.
// ---------------------------------------------------------------------------
#define BKP 40   // padded LDS row stride in bf16 elements

template <int STORE_BF16, int CAUSAL>
__global__ __launch_bounds__(256)
void gemm_nt_wmma(const __bf16* __restrict__ A, const __bf16* __restrict__ Wt,
                  void* __restrict__ Cout, int M, int N, int K,
                  long long bsA, long long bsB, long long bsC) {
    __shared__ __align__(16) __bf16 As[2][128 * BKP];
    __shared__ __align__(16) __bf16 Bs[2][128 * BKP];

    const int z = blockIdx.z;
    A  += (size_t)z * bsA;
    Wt += (size_t)z * bsB;
    float*  Cf = (float*)Cout  + (size_t)z * bsC;
    __bf16* Cb = (__bf16*)Cout + (size_t)z * bsC;

    const int bn0 = blockIdx.x * 128;
    const int bm0 = blockIdx.y * 128;

    const int tid    = threadIdx.x;
    const int wave   = tid >> 5;
    const int lane   = tid & 31;
    const int lane15 = lane & 15;
    const int hi     = lane >> 4;
    const int wrow   = (wave & 3) * 2;     // first of 2 tile rows
    const int wcol   = (wave >> 2) * 4;    // first of 4 tile cols

    // Fully-masked causal block: emit -inf and exit (block-uniform branch).
    if (CAUSAL && bn0 > bm0 + 127) {
#pragma unroll
        for (int i = 0; i < 2; ++i)
#pragma unroll
            for (int j = 0; j < 4; ++j) {
                int col = bn0 + (wcol + j) * 16 + lane15;
#pragma unroll
                for (int r = 0; r < 8; ++r) {
                    int row = bm0 + (wrow + i) * 16 + hi * 8 + r;
                    Cf[(size_t)row * N + col] = -INFINITY;  // CAUSAL => f32 store
                }
            }
        return;
    }

    const __bf16* Ablk = A  + (size_t)bm0 * K;
    const __bf16* Bblk = Wt + (size_t)bn0 * K;
    const int r0 = tid >> 2;          // 0..63
    const int sg = (tid & 3) * 8;     // 16B segment within 32-elem slab

    // stage one 128x32 A slab + 128x32 B slab into LDS buffer `b`
    auto stage = [&](int k0, int b) {
        const __bf16* ga = Ablk + k0 + sg;
        const __bf16* gb = Bblk + k0 + sg;
        __bf16* la = &As[b][0];
        __bf16* lb = &Bs[b][0];
        copy16_g2l(ga + (size_t)r0 * K,        la + r0 * BKP + sg);
        copy16_g2l(ga + (size_t)(r0 + 64) * K, la + (r0 + 64) * BKP + sg);
        copy16_g2l(gb + (size_t)r0 * K,        lb + r0 * BKP + sg);
        copy16_g2l(gb + (size_t)(r0 + 64) * K, lb + (r0 + 64) * BKP + sg);
        // speculative prefetch of the slab after this one -> global_prefetch_b8
        __builtin_prefetch(ga + (size_t)r0 * K + 32, 0, 0);
        __builtin_prefetch(gb + (size_t)r0 * K + 32, 0, 0);
    };

    v8f acc[2][4] = {};

    stage(0, 0);
    wait_g2l();
    __syncthreads();

    int buf = 0;
    for (int k0 = 0; k0 < K; k0 += 32) {
        if (k0 + 32 < K) stage(k0 + 32, buf ^ 1);   // overlap with compute

        // A fragments: 16x32 bf16, lanes 0-15 = M rows, K pairs per ISA layout
        v16bf af[2];
#pragma unroll
        for (int i = 0; i < 2; ++i) {
            const __bf16* Arow = &As[buf][((wrow + i) * 16 + lane15) * BKP];
#pragma unroll
            for (int p = 0; p < 8; ++p) {
                int kk = 2 * p + 8 * hi + ((p >= 4) ? 8 : 0);
                af[i][2 * p]     = Arow[kk];
                af[i][2 * p + 1] = Arow[kk + 1];
            }
        }
        // B fragments: 32x16 (KxN); lane = N, K pairs per ISA layout (same as A)
        v16bf bfr[4];
#pragma unroll
        for (int j = 0; j < 4; ++j) {
            const __bf16* Brow = &Bs[buf][((wcol + j) * 16 + lane15) * BKP];
#pragma unroll
            for (int p = 0; p < 8; ++p) {
                int kk = 2 * p + 8 * hi + ((p >= 4) ? 8 : 0);
                bfr[j][2 * p]     = Brow[kk];
                bfr[j][2 * p + 1] = Brow[kk + 1];
            }
        }
#pragma unroll
        for (int i = 0; i < 2; ++i)
#pragma unroll
            for (int j = 0; j < 4; ++j)
                acc[i][j] = __builtin_amdgcn_wmma_f32_16x16x32_bf16(
                    false, af[i], false, bfr[j], (short)0, acc[i][j], false, false);

        wait_g2l();        // next slab resident in LDS
        __syncthreads();   // all waves done reading `buf`, writes visible
        buf ^= 1;
    }

    // Epilogue: D layout -> VGPR r holds row (tile*16 + 8*hi + r), col = lane15
#pragma unroll
    for (int i = 0; i < 2; ++i)
#pragma unroll
        for (int j = 0; j < 4; ++j) {
            int col = bn0 + (wcol + j) * 16 + lane15;
#pragma unroll
            for (int r = 0; r < 8; ++r) {
                int row = bm0 + (wrow + i) * 16 + hi * 8 + r;
                float v = acc[i][j][r];
                if (CAUSAL && col > row) v = -INFINITY;
                size_t idx = (size_t)row * N + col;
                if (STORE_BF16) Cb[idx] = (__bf16)v;
                else            Cf[idx] = v;
            }
        }
}

// ---------------------------------------------------------------------------
// BatchNorm stats: per channel over 8192 rows, biased variance
// ---------------------------------------------------------------------------
__global__ __launch_bounds__(256) void bn_stats(const float* __restrict__ xa,
                                                float* __restrict__ mean,
                                                float* __restrict__ rsig) {
    int ch = blockIdx.x, tid = threadIdx.x;
    float s = 0.f, s2 = 0.f;
    for (int r = tid; r < ROWS; r += 256) {
        float v = xa[(size_t)r * C3 + ch];
        s += v; s2 += v * v;
    }
    __shared__ float sh1[256], sh2[256];
    sh1[tid] = s; sh2[tid] = s2;
    __syncthreads();
    for (int off = 128; off > 0; off >>= 1) {
        if (tid < off) { sh1[tid] += sh1[tid + off]; sh2[tid] += sh2[tid + off]; }
        __syncthreads();
    }
    if (tid == 0) {
        float m = sh1[0] * (1.f / ROWS);
        float var = sh2[0] * (1.f / ROWS) - m * m;
        mean[ch] = m;
        rsig[ch] = rsqrtf(var + BN_EPS);
    }
}

// ---------------------------------------------------------------------------
// BN apply + split into q (scaled by 1/sqrt(C)), k, and v transposed [b][c][t]
// ---------------------------------------------------------------------------
__global__ __launch_bounds__(256) void bn_apply_split(const float* __restrict__ xa,
                                                      const float* __restrict__ mean,
                                                      const float* __restrict__ rsig,
                                                      const float* __restrict__ gamma,
                                                      const float* __restrict__ beta,
                                                      __bf16* __restrict__ q,
                                                      __bf16* __restrict__ k,
                                                      __bf16* __restrict__ vt) {
    size_t i = (size_t)blockIdx.x * 256 + threadIdx.x;   // over 8192*3072
    int d = (int)(i % C3);
    size_t row = i / C3;                                  // b*T + t
    float val = (xa[i] - mean[d]) * rsig[d] * gamma[d] + beta[d];
    int c = d & (Csz - 1);
    int which = d >> 10;
    if (which == 0) {
        q[row * Csz + c] = (__bf16)(val * 0.03125f);      // fold 1/sqrt(1024)
    } else if (which == 1) {
        k[row * Csz + c] = (__bf16)val;
    } else {
        size_t b = row >> 10, t = row & (Tsz - 1);
        vt[(b * Csz + c) * Tsz + t] = (__bf16)val;        // transposed for P@V
    }
}

// ---------------------------------------------------------------------------
// Row softmax over 1024 cols (with -inf causal entries), bf16 output
// ---------------------------------------------------------------------------
__global__ __launch_bounds__(256) void softmax_rows(const float* __restrict__ S,
                                                    __bf16* __restrict__ P) {
    size_t row = blockIdx.x;
    int tid = threadIdx.x;
    const float* sr = S + row * Tsz;
    float v[4], m = -INFINITY;
#pragma unroll
    for (int i = 0; i < 4; ++i) { v[i] = sr[tid + i * 256]; m = fmaxf(m, v[i]); }
    __shared__ float sh[256];
    sh[tid] = m;
    __syncthreads();
    for (int off = 128; off > 0; off >>= 1) {
        if (tid < off) sh[tid] = fmaxf(sh[tid], sh[tid + off]);
        __syncthreads();
    }
    m = sh[0];
    __syncthreads();
    float s = 0.f;
#pragma unroll
    for (int i = 0; i < 4; ++i) { v[i] = __expf(v[i] - m); s += v[i]; }
    sh[tid] = s;
    __syncthreads();
    for (int off = 128; off > 0; off >>= 1) {
        if (tid < off) sh[tid] += sh[tid + off];
        __syncthreads();
    }
    float inv = 1.f / sh[0];
    __bf16* pr = P + row * Tsz;
#pragma unroll
    for (int i = 0; i < 4; ++i) pr[tid + i * 256] = (__bf16)(v[i] * inv);
}

// ---------------------------------------------------------------------------
// Orchestration
// ---------------------------------------------------------------------------
extern "C" void kernel_launch(void* const* d_in, const int* in_sizes, int n_in,
                              void* d_out, int out_size, void* d_ws, size_t ws_size,
                              hipStream_t stream) {
    (void)in_sizes; (void)n_in; (void)out_size; (void)ws_size;
    const float* x      = (const float*)d_in[0];
    const float* W_attn = (const float*)d_in[1];
    const float* W_proj = (const float*)d_in[2];
    const float* laA    = (const float*)d_in[3];
    const float* laB    = (const float*)d_in[4];
    const float* lpA    = (const float*)d_in[5];
    const float* lpB    = (const float*)d_in[6];
    const float* gamma  = (const float*)d_in[7];
    const float* beta   = (const float*)d_in[8];
    float* out = (float*)d_out;

    // --- workspace layout (256B aligned), with deliberate aliasing ----------
    char* base = (char*)d_ws;
    size_t off = 0;
    auto take = [&](size_t bytes) { size_t c = off; off = (off + bytes + 255) & ~(size_t)255; return c; };
    const size_t nX  = (size_t)ROWS * Csz;    // 8M
    const size_t nXA = (size_t)ROWS * C3;     // 25M

    __bf16* xb   = (__bf16*)(base + take(nX * 2));                 // x bf16
    __bf16* wma  = (__bf16*)(base + take((size_t)C3 * Csz * 2));   // merged attn W
    __bf16* wpb  = (__bf16*)(base + take((size_t)Csz * Csz * 2));  // W_proj bf16
    __bf16* wmp  = (__bf16*)(base + take((size_t)Csz * Csz * 2));  // merged proj W
    float*  xaF  = (float*)(base + take(nXA * 4));                 // xa f32 (96MB)
    float*  Sf   = xaF;                                            // alias: S (32MB)
    float*  mean = (float*)(base + take(C3 * 4));
    float*  rsig = (float*)(base + take(C3 * 4));
    __bf16* qb   = (__bf16*)(base + take(nX * 2));
    __bf16* Pb   = qb;                                             // alias after q used
    __bf16* kb   = (__bf16*)(base + take(nX * 2));
    __bf16* yb   = kb;                                             // alias after k used
    __bf16* vtb  = (__bf16*)(base + take(nX * 2));
    __bf16* y1b  = vtb;                                            // alias after v used

    // 1) precision conversions + LoRA merges
    cvt_f32_bf16<<<(int)(nX / 256), 256, 0, stream>>>(x, xb, (int)nX);
    cvt_f32_bf16<<<(Csz * Csz) / 256, 256, 0, stream>>>(W_proj, wpb, Csz * Csz);
    lora_merge<<<(C3 * Csz) / 256, 256, 0, stream>>>(W_attn, laA, laB, wma, C3 * Csz, C3);
    lora_merge<<<(Csz * Csz) / 256, 256, 0, stream>>>(W_proj, lpA, lpB, wmp, Csz * Csz, Csz);

    // 2) xa = x @ Wm_attn^T   [8192 x 3072 x 1024] -> f32
    gemm_nt_wmma<0, 0><<<dim3(C3 / 128, ROWS / 128, 1), 256, 0, stream>>>(
        xb, wma, xaF, ROWS, C3, Csz, 0, 0, 0);

    // 3) batchnorm stats + apply/split (q scaled, v transposed)
    bn_stats<<<C3, 256, 0, stream>>>(xaF, mean, rsig);
    bn_apply_split<<<(int)(nXA / 256), 256, 0, stream>>>(xaF, mean, rsig, gamma, beta,
                                                         qb, kb, vtb);

    // 4) S = q @ k^T (causal, per batch) -> f32 into alias of xa buffer
    gemm_nt_wmma<0, 1><<<dim3(Tsz / 128, Tsz / 128, Bsz), 256, 0, stream>>>(
        qb, kb, Sf, Tsz, Tsz, Csz,
        (long long)Tsz * Csz, (long long)Tsz * Csz, (long long)Tsz * Tsz);

    // 5) softmax rows -> bf16 P (aliases q)
    softmax_rows<<<(int)(Bsz * Tsz), 256, 0, stream>>>(Sf, Pb);

    // 6) y = P @ v  (v stored transposed -> NT form) -> bf16 (aliases k)
    gemm_nt_wmma<1, 0><<<dim3(Csz / 128, Tsz / 128, Bsz), 256, 0, stream>>>(
        Pb, vtb, yb, Tsz, Csz, Tsz,
        (long long)Tsz * Tsz, (long long)Csz * Tsz, (long long)Tsz * Csz);

    // 7) y1 = y @ W_proj^T -> bf16 (aliases vt)
    gemm_nt_wmma<1, 0><<<dim3(Csz / 128, ROWS / 128, 1), 256, 0, stream>>>(
        yb, wpb, y1b, ROWS, Csz, Csz, 0, 0, 0);

    // 8) out = y1 @ Wm_proj^T -> f32 directly into d_out
    gemm_nt_wmma<0, 0><<<dim3(Csz / 128, ROWS / 128, 1), 256, 0, stream>>>(
        y1b, wmp, out, ROWS, Csz, Csz, 0, 0, 0);
}
